// GATLayer_68238440399472
// MI455X (gfx1250) — matrix operand order repaired and checked by reading, not compiled, not run on hardware
//
#include <hip/hip_runtime.h>
#include <hip/hip_bf16.h>

typedef __attribute__((ext_vector_type(16))) _Float16 v16h;
typedef __attribute__((ext_vector_type(8)))  _Float16 v8h;
typedef __attribute__((ext_vector_type(8)))  float    v8f;
typedef __attribute__((ext_vector_type(8)))  int      v8i;

#define N_NODES 8192
#define F_DIM   512
#define D_DIM   64
#define LEAKY   0.2f

// ---------- kernel 0a: features f32 -> f16 (row-major) ----------
__global__ void k_cvt_feats(const float* __restrict__ f, _Float16* __restrict__ fh, int n) {
    int i = blockIdx.x * blockDim.x + threadIdx.x;
    if (i < n) fh[i] = (_Float16)f[i];
}

// ---------- kernel 0b: W[512][64] f32 -> WT_h[64][512] f16 (transposed) ----------
__global__ void k_cvt_wt(const float* __restrict__ W, _Float16* __restrict__ wt) {
    int i = blockIdx.x * blockDim.x + threadIdx.x;   // over 64*512
    if (i < D_DIM * F_DIM) {
        int d = i / F_DIM, k = i % F_DIM;
        wt[i] = (_Float16)W[k * D_DIM + d];
    }
}

// ---------- kernel 1: z = feats @ W via WMMA; emit zT (f16), s = z@a_self, t = z@a_neigh ----------
// One wave per 16-row tile of z. C tiles: 4 x (16x16 f32), K-loop over 512 in steps of 32.
__global__ void __launch_bounds__(32)
k_z_wmma(const _Float16* __restrict__ fh, const _Float16* __restrict__ wt,
         const float* __restrict__ a_self, const float* __restrict__ a_neigh,
         _Float16* __restrict__ zTh, float* __restrict__ s, float* __restrict__ t) {
    const int i0   = blockIdx.x * 16;      // 512 tiles
    const int lane = threadIdx.x & 31;
    const int n    = lane & 15;
    const int half = lane >> 4;

    v8f acc[4] = {};

    const _Float16* arow = fh + (size_t)(i0 + n) * F_DIM;
    for (int kk = 0; kk < F_DIM; kk += 32) {
        // A-fragment: feats rows i0..i0+15, K-slice kk..kk+31
        v8h alo = *(const v8h*)(arow + kk + 16 * half);        // e=0..7  -> k = e + 16*half
        v8h ahi = *(const v8h*)(arow + kk + 16 * half + 8);   // e=8..15 -> k = e + 8 + 16*half
        v16h af;
#pragma unroll
        for (int e = 0; e < 8; ++e) { af[e] = alo[e]; af[e + 8] = ahi[e]; }

#pragma unroll
        for (int nt = 0; nt < 4; ++nt) {
            // B-fragment: W K-slice x 16 cols, from transposed WT_h (contiguous per lane)
            v16h bf = *(const v16h*)(wt + (size_t)(nt * 16 + n) * F_DIM + kk + 16 * half);
            acc[nt] = __builtin_amdgcn_wmma_f32_16x16x32_f16(
                false, af, false, bf, (short)0, acc[nt], false, false);
        }
    }

    // Store z transposed as f16: zT[d][i]. C layout: vgpr k, lane -> row k+8*half, col nt*16+n
#pragma unroll
    for (int nt = 0; nt < 4; ++nt)
#pragma unroll
        for (int k = 0; k < 8; ++k)
            zTh[(size_t)(nt * 16 + n) * N_NODES + i0 + k + 8 * half] = (_Float16)acc[nt][k];

    // s, t per row via cross-lane reduction (lanes 0-15 and 16-31 hold disjoint row sets)
    float as[4], an[4];
#pragma unroll
    for (int nt = 0; nt < 4; ++nt) { as[nt] = a_self[nt * 16 + n]; an[nt] = a_neigh[nt * 16 + n]; }
#pragma unroll
    for (int k = 0; k < 8; ++k) {
        float ps = 0.f, pt = 0.f;
#pragma unroll
        for (int nt = 0; nt < 4; ++nt) { ps += acc[nt][k] * as[nt]; pt += acc[nt][k] * an[nt]; }
#pragma unroll
        for (int off = 8; off >= 1; off >>= 1) {   // masks < 16 keep halves separate
            ps += __shfl_xor(ps, off, 32);
            pt += __shfl_xor(pt, off, 32);
        }
        if (n == 0) {
            s[i0 + k + 8 * half] = ps;
            t[i0 + k + 8 * half] = pt;
        }
    }
}

// ---------- kernel 2: fused masked-softmax + attn@z, streaming over j ----------
// 512 workgroups x 256 threads; each of 8 waves handles a 1024-column j-chunk of
// the same 16-row tile; partial (acc, l) combined in LDS; h = acc / l.
__global__ void __launch_bounds__(256)
k_attn(const int* __restrict__ A, const _Float16* __restrict__ zTh,
       const float* __restrict__ s, const float* __restrict__ t,
       float* __restrict__ out) {
    __shared__ float sacc[16 * 64];
    __shared__ float sl[16];
    const int tid = threadIdx.x;
    for (int idx = tid; idx < 16 * 64; idx += 256) sacc[idx] = 0.f;
    if (tid < 16) sl[tid] = 0.f;
    __syncthreads();

    const int i0   = blockIdx.x * 16;
    const int lane = tid & 31;
    const int wave = tid >> 5;          // 0..7
    const int n    = lane & 15;
    const int half = lane >> 4;

    const float s_m = s[i0 + n];        // this lane's row score

    v8f acc[4] = {};
    v8f accl   = {};
    v16h ones;
#pragma unroll
    for (int e = 0; e < 16; ++e) ones[e] = (_Float16)1.0f;

    const int* arow = A + (size_t)(i0 + n) * N_NODES;
    const int jbase = wave * 1024;

    for (int it = 0; it < 32; ++it) {
        const int j0 = jbase + it * 32;
        v8i A1 = *(const v8i*)(arow + j0 + 16 * half);         // k = e + 16*half
        v8i A2 = *(const v8i*)(arow + j0 + 16 * half + 8);    // k = e + 8 + 16*half
        v8f t1 = *(const v8f*)(t + j0 + 16 * half);
        v8f t2 = *(const v8f*)(t + j0 + 16 * half + 8);

        // P in A-operand layout: p = edge ? exp(leaky(s_i + t_j)) : 0
        v16h af;
#pragma unroll
        for (int e = 0; e < 8; ++e) {
            float x1 = s_m + t1[e]; x1 = (x1 > 0.f) ? x1 : LEAKY * x1;
            float p1 = (A1[e] > 0) ? __expf(x1) : 0.f;
            af[e] = (_Float16)p1;
            float x2 = s_m + t2[e]; x2 = (x2 > 0.f) ? x2 : LEAKY * x2;
            float p2 = (A2[e] > 0) ? __expf(x2) : 0.f;
            af[e + 8] = (_Float16)p2;
        }

        // acc += P (16x32) @ Z (32x64) as 4 WMMA n-tiles; B frags contiguous from zT
#pragma unroll
        for (int nt = 0; nt < 4; ++nt) {
            v16h bf = *(const v16h*)(zTh + (size_t)(nt * 16 + n) * N_NODES + j0 + 16 * half);
            acc[nt] = __builtin_amdgcn_wmma_f32_16x16x32_f16(
                false, af, false, bf, (short)0, acc[nt], false, false);
        }
        // Row sums: P @ ones -> every column of C holds the row sum
        accl = __builtin_amdgcn_wmma_f32_16x16x32_f16(
            false, af, false, ones, (short)0, accl, false, false);
    }

    // Combine wave partials in LDS. C layout: vgpr k, lane -> row k+8*half, col nt*16+n
#pragma unroll
    for (int nt = 0; nt < 4; ++nt)
#pragma unroll
        for (int k = 0; k < 8; ++k)
            atomicAdd(&sacc[(k + 8 * half) * 64 + nt * 16 + n], acc[nt][k]);
    if (n == 0) {
#pragma unroll
        for (int k = 0; k < 8; ++k)
            atomicAdd(&sl[k + 8 * half], accl[k]);
    }
    __syncthreads();

    // h = acc / l
    for (int idx = tid; idx < 16 * 64; idx += 256) {
        int r = idx >> 6;
        out[(size_t)(i0 + r) * 64 + (idx & 63)] = sacc[idx] / sl[r];
    }
}

extern "C" void kernel_launch(void* const* d_in, const int* in_sizes, int n_in,
                              void* d_out, int out_size, void* d_ws, size_t ws_size,
                              hipStream_t stream) {
    const float* features = (const float*)d_in[0];   // [8192,512] f32
    const int*   A        = (const int*)d_in[1];     // [8192,8192] i32
    const float* W        = (const float*)d_in[2];   // [512,64] f32
    const float* a_self   = (const float*)d_in[3];   // [64,1] f32
    const float* a_neigh  = (const float*)d_in[4];   // [64,1] f32
    float* out = (float*)d_out;                      // [8192,64] f32

    char* ws = (char*)d_ws;                          // ~9.6 MB used
    _Float16* fh  = (_Float16*)(ws + 0);             // 8192*512*2 = 8388608
    _Float16* wt  = (_Float16*)(ws + 8388608);       // 64*512*2   = 65536
    _Float16* zTh = (_Float16*)(ws + 8454144);       // 64*8192*2  = 1048576
    float*    s   = (float*)(ws + 9502720);          // 8192*4
    float*    t   = (float*)(ws + 9535488);          // 8192*4

    k_cvt_feats<<<(N_NODES * F_DIM + 255) / 256, 256, 0, stream>>>(features, fh, N_NODES * F_DIM);
    k_cvt_wt<<<(D_DIM * F_DIM + 255) / 256, 256, 0, stream>>>(W, wt);
    k_z_wmma<<<512, 32, 0, stream>>>(fh, wt, a_self, a_neigh, zTh, s, t);
    k_attn<<<512, 256, 0, stream>>>(A, zTh, s, t, out);
}
